// MSFLMamba_51539607684
// MI455X (gfx1250) — matrix operand order repaired and checked by reading, not compiled, not run on hardware
//
#include <hip/hip_runtime.h>
#include <cmath>

typedef __bf16 bf16_t;
typedef __bf16 v8bf  __attribute__((ext_vector_type(8)));
typedef __bf16 v16bf __attribute__((ext_vector_type(16)));
typedef float  v2f   __attribute__((ext_vector_type(2)));
typedef float  v8f   __attribute__((ext_vector_type(8)));

#define BDIM  256
#define CH    256
#define CH2   512
#define HH    80
#define WW    80
#define HWPIX 6400
#define NB    8

__device__ __forceinline__ float gelu_exact(float v) {
    return 0.5f * v * (1.0f + erff(v * 0.70710678118654752f));
}

// ---------------------------------------------------------------------------
// K0: build the 80x80 spatial operator Hm^T for irfft(i * rfft(h)) (ortho).
// Stored transposed: HmT[m*80+n] = -(2/N) * sum_{j=1}^{N/2-1} sin(2*pi*j*(n-m)/N)
// ---------------------------------------------------------------------------
__global__ __launch_bounds__(256) void k_hmat(float* __restrict__ Hm) {
    for (int i = threadIdx.x; i < HWPIX; i += BDIM) {
        int m = i / WW;
        int n = i - m * WW;
        int d = n - m;
        float s = 0.f;
        for (int j = 1; j <= WW / 2 - 1; ++j)
            s += sinf(6.283185307179586f * (float)(j * d) / (float)WW);
        Hm[i] = -(2.0f / (float)WW) * s;
    }
}

// ---------------------------------------------------------------------------
// K1: BatchNorm (inference) + spectral gating reduced to a 1-D op along W:
//     s = alpha_r * h + alpha_i * (h @ HmT)   (per 80x80 plane)
// The 80x80 @ 80x80 product runs on V_WMMA_F32_16X16X4_F32 (exact fp32):
// 25 tiles of 16x16 per plane, K=80 in steps of 4. One block per (b,c) plane.
// ---------------------------------------------------------------------------
__global__ __launch_bounds__(256) void k_bn_spectral(
    const float* __restrict__ x, const float* __restrict__ gma,
    const float* __restrict__ bta, const float* __restrict__ mn,
    const float* __restrict__ vr, const float* __restrict__ ar,
    const float* __restrict__ ai, const float* __restrict__ Hm,
    float* __restrict__ s_f32, bf16_t* __restrict__ s_bf)
{
    __shared__ float hs[HWPIX];   // BN output, [r][m] row-major
    __shared__ float Ht[HWPIX];   // HmT, [m][n] row-major
    const int tid = threadIdx.x;
    const int plane = blockIdx.x;                // b*C + c
    const int c = plane & (CH - 1);
    const float sc = gma[c] * rsqrtf(vr[c] + 1e-5f);
    const float sh = bta[c] - mn[c] * sc;
    const float ga = ar[c], gb = ai[c];
    const long base = (long)plane * HWPIX;
    for (int i = tid; i < HWPIX; i += BDIM) {
        hs[i] = x[base + i] * sc + sh;
        Ht[i] = Hm[i];
    }
    __syncthreads();

    const int wv = tid >> 5;
    const int lane = tid & 31;
    const int l = lane & 15;
    const int hi = lane >> 4;
    // 25 output tiles distributed round-robin over the 8 waves
    for (int t = wv; t < 25; t += 8) {
        const int tm = t / 5;
        const int tn = t - tm * 5;
        const int mBase = tm * 16, nBase = tn * 16;
        v8f acc;
        #pragma unroll
        for (int r = 0; r < 8; ++r) acc[r] = 0.f;
        #pragma unroll 4
        for (int k = 0; k < WW; k += 4) {
            // A frag (16x4 fp32): lane<16 -> K=k,k+1 ; lane>=16 -> K=k+2,k+3
            const v2f a = *reinterpret_cast<const v2f*>(
                &hs[(mBase + l) * WW + k + 2 * hi]);
            v2f b;
            b[0] = Ht[(k + 2 * hi) * WW + nBase + l];
            b[1] = Ht[(k + 2 * hi + 1) * WW + nBase + l];
            acc = __builtin_amdgcn_wmma_f32_16x16x4_f32(
                false, a, false, b, (short)0, acc, false, false);
        }
        // C/D 16x16: lane<16 -> M=mBase+r ; lane>=16 -> M=mBase+8+r ; N=nBase+l
        #pragma unroll
        for (int r = 0; r < 8; ++r) {
            const int m = mBase + hi * 8 + r;
            const int idx = m * WW + nBase + l;
            const float v = ga * hs[idx] + gb * acc[r];
            s_f32[base + idx] = v;
            s_bf[base + idx] = (bf16_t)v;
        }
    }
}

// ---------------------------------------------------------------------------
// K2: depthwise 3x3 (SAME, zero pad) + bias + exact GELU, bf16 in/out.
// ---------------------------------------------------------------------------
__global__ __launch_bounds__(256) void k_dwconv_gelu(
    const bf16_t* __restrict__ in, const float* __restrict__ w9,
    const float* __restrict__ bia, bf16_t* __restrict__ out)
{
    const long idx = (long)blockIdx.x * BDIM + threadIdx.x; // total exact multiple
    const int p = (int)(idx % HWPIX);
    const long plane = idx / HWPIX;
    const int c = (int)(plane & (CH - 1));
    const int ph = p / WW, pw = p - ph * WW;
    float acc = bia[c];
    #pragma unroll
    for (int kh = 0; kh < 3; ++kh) {
        const int hh = ph + kh - 1;
        if ((unsigned)hh >= HH) continue;
        #pragma unroll
        for (int kw = 0; kw < 3; ++kw) {
            const int ww2 = pw + kw - 1;
            if ((unsigned)ww2 >= WW) continue;
            acc += (float)in[plane * HWPIX + hh * WW + ww2] * w9[c * 9 + kh * 3 + kw];
        }
    }
    out[idx] = (bf16_t)gelu_exact(acc);
}

// ---------------------------------------------------------------------------
// WMMA fragment helper: combine two contiguous 8x-bf16 LDS chunks (b128 loads)
// ---------------------------------------------------------------------------
__device__ __forceinline__ v16bf mk16(const bf16_t* p0, const bf16_t* p1) {
    v8bf a = *reinterpret_cast<const v8bf*>(p0);
    v8bf b = *reinterpret_cast<const v8bf*>(p1);
    v16bf r;
    #pragma unroll
    for (int i = 0; i < 8; ++i) { r[i] = a[i]; r[i + 8] = b[i]; }
    return r;
}

// ---------------------------------------------------------------------------
// K3: generic 1x1-conv GEMM:  out[co,p] = sum_ci W[co,ci]*in[ci,p] + bias[co]
// Block tile 128(cout) x 128(pixels), K-step 32, double-buffered LDS.
// 8 waves, each a 64x32 patch => 8 v8f accumulators, v_wmma_f32_16x16x32_bf16.
// Epilogue: LDS bias, batched residual loads, optional exact-GELU.
// ---------------------------------------------------------------------------
__global__ __launch_bounds__(256) void k_gemm(
    const bf16_t* __restrict__ in, const float* __restrict__ wgt,
    const float* __restrict__ bia, bf16_t* __restrict__ out_bf,
    float* __restrict__ out_f32, const float* __restrict__ res,
    int Cin, int Cout, int do_gelu)
{
    __shared__ bf16_t As[2][128 * 40];   // weights  [co][k], padded stride 40
    __shared__ bf16_t Bs[2][128 * 40];   // acts     [p][k],  padded stride 40
    __shared__ float  biasS[128];
    const int tid  = threadIdx.x;
    const int wv   = tid >> 5;
    const int lane = tid & 31;
    const int l    = lane & 15;
    const int hi   = lane >> 4;
    const int mWave = (wv >> 2) * 64;   // 2 waves along M
    const int nWave = (wv & 3) * 32;    // 4 waves along N
    const int pBase  = blockIdx.x * 128;
    const int coBase = blockIdx.y * 128;
    const long inB  = (long)blockIdx.z * Cin  * HWPIX;
    const long outB = (long)blockIdx.z * Cout * HWPIX;

    if (tid < 128) biasS[tid] = bia[coBase + tid];

    v8f acc[4][2];
    #pragma unroll
    for (int i = 0; i < 4; ++i)
        #pragma unroll
        for (int j = 0; j < 2; ++j)
            #pragma unroll
            for (int r = 0; r < 8; ++r) acc[i][j][r] = 0.f;

    const int kSteps = Cin >> 5;

    // global -> LDS tile loader (stage `buf`, k-step `ks`)
    auto loadTile = [&](int buf, int ks) {
        const int kBase = ks << 5;
        // A: 128x32 fp32 weights -> bf16 (4 float4 per thread)
        #pragma unroll
        for (int i = 0; i < 4; ++i) {
            const int idx4 = tid + i * 256;
            const int row = idx4 >> 3;
            const int c4 = (idx4 & 7) << 2;
            const float4 f = *reinterpret_cast<const float4*>(
                &wgt[(long)(coBase + row) * Cin + kBase + c4]);
            bf16_t* dst = &As[buf][row * 40 + c4];
            dst[0] = (bf16_t)f.x; dst[1] = (bf16_t)f.y;
            dst[2] = (bf16_t)f.z; dst[3] = (bf16_t)f.w;
        }
        // B: 32(k) x 128(p) bf16 from global, transposed into LDS as [p][k]
        const int k0 = (tid >> 4) << 1;
        const int p0 = (tid & 15) << 3;
        #pragma unroll
        for (int dk = 0; dk < 2; ++dk) {
            const v8bf v = *reinterpret_cast<const v8bf*>(
                &in[inB + (long)(kBase + k0 + dk) * HWPIX + pBase + p0]);
            #pragma unroll
            for (int j = 0; j < 8; ++j) Bs[buf][(p0 + j) * 40 + k0 + dk] = v[j];
        }
    };

    loadTile(0, 0);
    __syncthreads();
    for (int ks = 0; ks < kSteps; ++ks) {
        const int cur = ks & 1;
        if (ks + 1 < kSteps) loadTile(cur ^ 1, ks + 1);
        // fragments per the ISA 16-bit A(16x32)/B(32x16) lane layouts
        v16bf af[4], bfr[2];
        #pragma unroll
        for (int i = 0; i < 4; ++i) {
            const int row = mWave + i * 16 + l;
            af[i] = mk16(&As[cur][row * 40 + hi * 8],
                         &As[cur][row * 40 + 16 + hi * 8]);
        }
        #pragma unroll
        for (int j = 0; j < 2; ++j) {
            const int col = nWave + j * 16 + l;
            bfr[j] = mk16(&Bs[cur][col * 40 + hi * 8],
                          &Bs[cur][col * 40 + 16 + hi * 8]);
        }
        #pragma unroll
        for (int i = 0; i < 4; ++i)
            #pragma unroll
            for (int j = 0; j < 2; ++j)
                acc[i][j] = __builtin_amdgcn_wmma_f32_16x16x32_bf16(
                    false, af[i], false, bfr[j], (short)0, acc[i][j], false, false);
        __syncthreads();
    }

    // epilogue: bias from LDS, batched residual loads, (+gelu), stores
    #pragma unroll
    for (int i = 0; i < 4; ++i) {
        #pragma unroll
        for (int j = 0; j < 2; ++j) {
            const int col = pBase + nWave + j * 16 + l;
            const int mtl = mWave + i * 16 + hi * 8;
            float rv[8];
            if (res) {
                #pragma unroll
                for (int r = 0; r < 8; ++r)
                    rv[r] = res[outB + (long)(coBase + mtl + r) * HWPIX + col];
            }
            #pragma unroll
            for (int r = 0; r < 8; ++r) {
                const int m = mtl + r;
                float v = acc[i][j][r] + biasS[m];
                if (do_gelu) v = gelu_exact(v);
                if (res) v += rv[r];
                const long oidx = outB + (long)(coBase + m) * HWPIX + col;
                if (out_bf)  out_bf[oidx]  = (bf16_t)v;
                if (out_f32) out_f32[oidx] = v;
            }
        }
    }
}

// ---------------------------------------------------------------------------
extern "C" void kernel_launch(void* const* d_in, const int* in_sizes, int n_in,
                              void* d_out, int out_size, void* d_ws, size_t ws_size,
                              hipStream_t stream)
{
    (void)in_sizes; (void)n_in; (void)out_size; (void)ws_size;
    const float* x    = (const float*)d_in[0];
    const float* g    = (const float*)d_in[1];
    const float* be   = (const float*)d_in[2];
    const float* mn   = (const float*)d_in[3];
    const float* vr   = (const float*)d_in[4];
    const float* ar   = (const float*)d_in[5];
    const float* ai   = (const float*)d_in[6];
    const float* pi_w = (const float*)d_in[7];
    const float* pi_b = (const float*)d_in[8];
    const float* dw_w = (const float*)d_in[9];
    const float* dw_b = (const float*)d_in[10];
    const float* fp_w = (const float*)d_in[11];
    const float* fp_b = (const float*)d_in[12];
    const float* f1_w = (const float*)d_in[13];
    const float* f1_b = (const float*)d_in[14];
    const float* f2_w = (const float*)d_in[15];
    const float* f2_b = (const float*)d_in[16];
    const float* po_w = (const float*)d_in[17];
    const float* po_b = (const float*)d_in[18];
    const float* ps_w = (const float*)d_in[19];
    const float* ps_b = (const float*)d_in[20];
    float* out = (float*)d_out;

    char* ws = (char*)d_ws;
    size_t off = 0;
    float*  Hm   = (float*)(ws + off);  off += (size_t)HWPIX * 4;
    float*  sF   = (float*)(ws + off);  off += (size_t)NB * CH  * HWPIX * 4;
    bf16_t* bufA = (bf16_t*)(ws + off); off += (size_t)NB * CH  * HWPIX * 2;
    bf16_t* bufB = (bf16_t*)(ws + off); off += (size_t)NB * CH  * HWPIX * 2;
    bf16_t* bufF = (bf16_t*)(ws + off); off += (size_t)NB * CH2 * HWPIX * 2;

    const dim3 blk(BDIM);
    const dim3 g256(HWPIX / 128, CH  / 128, NB);   // (50, 2, 8)
    const dim3 g512(HWPIX / 128, CH2 / 128, NB);   // (50, 4, 8)

    // spectral operator + BN + gating  -> sF (fp32 residual), bufA (bf16)
    k_hmat<<<1, blk, 0, stream>>>(Hm);
    k_bn_spectral<<<NB * CH, blk, 0, stream>>>(x, g, be, mn, vr, ar, ai, Hm, sF, bufA);
    // proj_in (1x1) : bufA -> bufB
    k_gemm<<<g256, blk, 0, stream>>>(bufA, pi_w, pi_b, bufB, nullptr, nullptr, CH, CH, 0);
    // depthwise 3x3 + GELU : bufB -> bufA
    k_dwconv_gelu<<<(NB * CH * HWPIX) / BDIM, blk, 0, stream>>>(bufB, dw_w, dw_b, bufA);
    // fpw (1x1) : bufA -> bufB
    k_gemm<<<g256, blk, 0, stream>>>(bufA, fp_w, fp_b, bufB, nullptr, nullptr, CH, CH, 0);
    // ffn1 (256->512) + GELU : bufB -> bufF
    k_gemm<<<g512, blk, 0, stream>>>(bufB, f1_w, f1_b, bufF, nullptr, nullptr, CH, CH2, 1);
    // ffn2 (512->256) : bufF -> bufA
    k_gemm<<<g256, blk, 0, stream>>>(bufF, f2_w, f2_b, bufA, nullptr, nullptr, CH2, CH, 0);
    // proj_out (1x1) + res2(sF) : bufA -> bufB
    k_gemm<<<g256, blk, 0, stream>>>(bufA, po_w, po_b, bufB, nullptr, sF, CH, CH, 0);
    // post (1x1) + outer residual(x) : bufB -> d_out (fp32)
    k_gemm<<<g256, blk, 0, stream>>>(bufB, ps_w, ps_b, nullptr, out, x, CH, CH, 0);
}
